// SAModule_83219286327860
// MI455X (gfx1250) — matrix operand order repaired and checked by reading, not compiled, not run on hardware
//
#include <hip/hip_runtime.h>

// ---------------------------------------------------------------------------
// Problem constants (from reference)
// ---------------------------------------------------------------------------
#define NPTS 8192
#define KNBR 64
#define FIN 64
#define DE 4
#define HID 128
#define FOUT 128
#define EG 262144
#define R2_CUT 4.0f

// points per block in the MLP kernel (M = PPB*64 rows)
#define PPB 2
#define MROWS (PPB * KNBR)          // 128
#define MTILES (MROWS / 16)         // 8

// hash table
#define HBITS 20
#define HSZ (1u << HBITS)
#define HEMPTY 0xFFFFFFFFu

// workspace layout (bytes, all 256-aligned)
#define OFF_NBR    0u            // N*K int            = 2097152
#define OFF_VALID  2097152u      // N*K u8             =  524288
#define OFF_EIDX   2621440u      // N*K int            = 2097152
#define OFF_HKEY   4718592u      // HSZ u32            = 4194304
#define OFF_HVAL   8912896u      // HSZ u32            = 4194304
#define OFF_W1T    13107200u     // 128*96  bf16       =   24576
#define OFF_W2T    13131776u     // 128*128 bf16       =   32768
#define OFF_WGT    13164544u     // 128*128 bf16       =   32768
#define OFF_AGG    13197312u     // N*128 f32          = 4194304
// total ~ 17.4 MB

typedef __attribute__((ext_vector_type(16))) __bf16 v16bf;
typedef __attribute__((ext_vector_type(8)))  float  v8f;

union BF16x16 { v16bf v; unsigned int u[8]; };
union F32x8   { v8f  v; float f[8]; };

__device__ __forceinline__ unsigned short f2bf(float f) {
    unsigned u = __float_as_uint(f);
    unsigned r = u + 0x7FFFu + ((u >> 16) & 1u);   // round-to-nearest-even
    return (unsigned short)(r >> 16);
}

// A fragment: 16x32 bf16 tile, row-major source (stride in elements).
// ISA layout: lanes 0-15 row m, VGPR v<4 -> K=2v,2v+1 ; v>=4 -> K=16+2(v-4);
// lanes 16-31 same rows, K offset +8 in each half.
__device__ __forceinline__ v16bf load_a_frag(const unsigned short* p, int row,
                                             int stride, int k0, int hi) {
    BF16x16 t;
    const unsigned short* rp = p + row * stride + k0;
#pragma unroll
    for (int v = 0; v < 8; ++v) {
        int koff = ((v & 4) << 2) + (hi << 3) + ((v & 3) << 1);
        t.u[v] = *(const unsigned int*)(rp + koff);
    }
    return t.v;
}

// B fragment: 32x16 bf16 tile from K-major (transposed) weights.
// same K layout as A: VGPR v<4 -> K=2v,2v+1 ; v>=4 -> K=16+2(v-4); lanes 16-31 K offset +8.
__device__ __forceinline__ v16bf load_b_frag(const unsigned short* bt, int col,
                                             int stride, int k0, int hi) {
    BF16x16 t;
    const unsigned short* cp = bt + col * stride + k0;
#pragma unroll
    for (int v = 0; v < 8; ++v) {
        int koff = ((v & 4) << 2) + (hi << 3) + ((v & 3) << 1);
        t.u[v] = *(const unsigned int*)(cp + koff);
    }
    return t.v;
}

// ---------------------------------------------------------------------------
// 0) weight transpose + bf16 convert
// ---------------------------------------------------------------------------
__global__ __launch_bounds__(256) void prep_weights(
    const float* __restrict__ W1, const float* __restrict__ W2,
    const float* __restrict__ Wg,
    unsigned short* __restrict__ W1T, unsigned short* __restrict__ W2T,
    unsigned short* __restrict__ WgT) {
    int tid = threadIdx.x;
    for (int t = tid; t < HID * 96; t += 256) {       // W1: [72,128] -> [128][96] pad
        int n = t / 96, k = t % 96;
        float v = (k < FIN + DE + 4) ? W1[k * HID + n] : 0.0f;
        W1T[t] = f2bf(v);
    }
    for (int t = tid; t < HID * HID; t += 256) {      // W2: [128,128] -> [128][128]
        int n = t / HID, k = t % HID;
        W2T[t] = f2bf(W2[k * HID + n]);
    }
    for (int t = tid; t < HID * FOUT; t += 256) {     // Wg
        int n = t / HID, k = t % HID;
        WgT[t] = f2bf(Wg[k * FOUT + n]);
    }
}

// ---------------------------------------------------------------------------
// 1) hash table for global edge lookup
// ---------------------------------------------------------------------------
__global__ __launch_bounds__(256) void init_hash(unsigned int* hkey, unsigned int* hval) {
    unsigned t = blockIdx.x * 256u + threadIdx.x;
    if (t < HSZ) { hkey[t] = HEMPTY; hval[t] = 0x7FFFFFFFu; }
}

__global__ __launch_bounds__(256) void build_hash(const int* __restrict__ eidx2,
                                                  unsigned int* hkey, unsigned int* hval) {
    int e = blockIdx.x * 256 + threadIdx.x;
    if (e >= EG) return;
    int j = eidx2[e];            // edge_index[0][e]  (source)
    int i = eidx2[EG + e];       // edge_index[1][e]  (target)
    unsigned key = (unsigned)j * (unsigned)NPTS + (unsigned)i;
    unsigned s = (key * 2654435761u) >> (32 - HBITS);
    for (unsigned p = 0; p < HSZ; ++p) {
        unsigned old = atomicCAS(&hkey[s], HEMPTY, key);
        if (old == HEMPTY || old == key) {
            atomicMin((int*)&hval[s], e);   // leftmost (smallest id) among dups
            return;
        }
        s = (s + 1) & (HSZ - 1);
    }
}

// ---------------------------------------------------------------------------
// 2) radius top-K neighbor search (one block per target point)
// ---------------------------------------------------------------------------
#define CAND 1024

__global__ __launch_bounds__(256) void neighbors_kernel(
    const float* __restrict__ pos, const int* __restrict__ batch,
    int* __restrict__ nbr, unsigned char* __restrict__ valid) {
    __shared__ int s_lo, s_hi, s_cnt;
    __shared__ float cd2[CAND];
    __shared__ int   cid[CAND];

    int i = blockIdx.x, tid = threadIdx.x;
    float px = pos[i * 3 + 0], py = pos[i * 3 + 1], pz = pos[i * 3 + 2];

    if (tid == 0) {
        int b = batch[i];
        int lo = 0, hi = NPTS;
        while (lo < hi) { int m = (lo + hi) >> 1; if (batch[m] < b) lo = m + 1; else hi = m; }
        s_lo = lo;
        int lo2 = 0, hi2 = NPTS;
        while (lo2 < hi2) { int m = (lo2 + hi2) >> 1; if (batch[m] <= b) lo2 = m + 1; else hi2 = m; }
        s_hi = lo2;
        s_cnt = 0;
    }
    __syncthreads();

    for (int j = s_lo + tid; j < s_hi; j += 256) {
        float dx = pos[j * 3 + 0] - px, dy = pos[j * 3 + 1] - py, dz = pos[j * 3 + 2] - pz;
        float d2 = dx * dx + dy * dy + dz * dz;
        if (d2 <= R2_CUT) {
            int slot = atomicAdd(&s_cnt, 1);
            if (slot < CAND) { cd2[slot] = d2; cid[slot] = j; }
        }
    }
    __syncthreads();
    int cnt = min(s_cnt, CAND);
    for (int t = cnt + tid; t < CAND; t += 256) { cd2[t] = 3.0e38f; cid[t] = 0x7FFFFFFF; }
    __syncthreads();

    // bitonic sort ascending by (d2, idx) -- deterministic tie-break
    for (unsigned k = 2; k <= CAND; k <<= 1) {
        for (unsigned j = k >> 1; j > 0; j >>= 1) {
            for (unsigned t = tid; t < CAND; t += 256) {
                unsigned l = t ^ j;
                if (l > t) {
                    float a = cd2[t], b = cd2[l];
                    int ia = cid[t], ib = cid[l];
                    bool gt = (a > b) || (a == b && ia > ib);
                    bool asc = ((t & k) == 0);
                    if (gt == asc) { cd2[t] = b; cd2[l] = a; cid[t] = ib; cid[l] = ia; }
                }
            }
            __syncthreads();
        }
    }
    if (tid < KNBR) {
        bool ok = tid < cnt;
        nbr[i * KNBR + tid]   = ok ? cid[tid] : i;
        valid[i * KNBR + tid] = ok ? 1 : 0;
    }
}

// ---------------------------------------------------------------------------
// 3) edge-attr lookup via hash probe
// ---------------------------------------------------------------------------
__global__ __launch_bounds__(256) void edge_lookup(
    const int* __restrict__ nbr, int* __restrict__ eidx,
    const unsigned int* __restrict__ hkey, const unsigned int* __restrict__ hval) {
    int t = blockIdx.x * 256 + threadIdx.x;
    if (t >= NPTS * KNBR) return;
    int i = t >> 6;
    int j = nbr[t];
    unsigned key = (unsigned)j * (unsigned)NPTS + (unsigned)i;
    unsigned s = (key * 2654435761u) >> (32 - HBITS);
    int found = EG;   // sentinel row
    for (unsigned p = 0; p < HSZ; ++p) {
        unsigned hk = hkey[s];
        if (hk == key) { found = (int)hval[s]; break; }
        if (hk == HEMPTY) break;
        s = (s + 1) & (HSZ - 1);
    }
    eidx[t] = found;
}

// ---------------------------------------------------------------------------
// 4) message MLP + masked max-aggregate (WMMA bf16)
//    one block (8 waves / 256 threads) per PPB=2 target points; M=128 rows
// ---------------------------------------------------------------------------
__device__ __forceinline__ float angle3(float ax, float ay, float az,
                                        float bx, float by, float bz) {
    float cx = ay * bz - az * by;
    float cy = az * bx - ax * bz;
    float cz = ax * by - ay * bx;
    float cn2 = cx * cx + cy * cy + cz * cz;
    float cn = cn2 > 0.0f ? __builtin_sqrtf(cn2) : 0.0f;
    float d = ax * bx + ay * by + az * bz;
    if (cn == 0.0f && d == 0.0f) d = 1.0f;   // atan2(0,1)=0
    return atan2f(cn, d);
}

__global__ __launch_bounds__(256) void mlp_kernel(
    const float* __restrict__ x, const float* __restrict__ pos,
    const float* __restrict__ normal, const float* __restrict__ eattr,
    const int* __restrict__ nbr, const unsigned char* __restrict__ valid,
    const int* __restrict__ eidx,
    const unsigned short* __restrict__ W1T, const unsigned short* __restrict__ W2T,
    const float* __restrict__ b1, const float* __restrict__ b2,
    float* __restrict__ agg) {
    extern __shared__ unsigned char smem[];
    unsigned short* w1s  = (unsigned short*)smem;                       // 128*96
    unsigned short* w2s  = w1s + HID * 96;                              // 128*128
    unsigned short* msg  = w2s + HID * HID;                             // MROWS*96
    unsigned short* hb   = msg + MROWS * 96;                            // MROWS*128
    float* ppf           = (float*)(hb + MROWS * HID);                  // MROWS*4
    float* bias1         = ppf + MROWS * 4;                             // 128
    float* bias2         = bias1 + HID;                                 // 128
    int*   jrow          = (int*)(bias2 + HID);                         // MROWS
    int*   vm            = jrow + MROWS;                                // MROWS
    int*   erow          = vm + MROWS;                                  // MROWS

    int i0 = blockIdx.x * PPB, tid = threadIdx.x;
    int wv = tid >> 5, lane = tid & 31, lo16 = lane & 15, hi = lane >> 4;

    // get weight cachelines moving early (global_prefetch_b8)
    __builtin_prefetch(W1T + tid * 64, 0, 1);
    __builtin_prefetch(W2T + tid * 64, 0, 1);

    // stage weights + biases into LDS (dword copies, global is L2-resident)
    for (int t = tid; t < HID * 96 / 2; t += 256)
        ((unsigned int*)w1s)[t] = ((const unsigned int*)W1T)[t];
    for (int t = tid; t < HID * HID / 2; t += 256)
        ((unsigned int*)w2s)[t] = ((const unsigned int*)W2T)[t];
    if (tid < HID) { bias1[tid] = b1[tid]; bias2[tid] = b2[tid]; }

    // per-message metadata + point-pair features (row m -> target i0+m/64, k=m%64)
    if (tid < MROWS) {
        int m = tid;
        int i = i0 + (m >> 6);
        int k = m & (KNBR - 1);
        int j = nbr[i * KNBR + k];
        jrow[m] = j;
        vm[m]   = valid[i * KNBR + k];
        erow[m] = eidx[i * KNBR + k];
        float dx = pos[j * 3 + 0] - pos[i * 3 + 0];
        float dy = pos[j * 3 + 1] - pos[i * 3 + 1];
        float dz = pos[j * 3 + 2] - pos[i * 3 + 2];
        float nix = normal[i * 3 + 0], niy = normal[i * 3 + 1], niz = normal[i * 3 + 2];
        float njx = normal[j * 3 + 0], njy = normal[j * 3 + 1], njz = normal[j * 3 + 2];
        float d2 = dx * dx + dy * dy + dz * dz;
        ppf[m * 4 + 0] = d2 > 0.0f ? __builtin_sqrtf(d2) : 0.0f;
        ppf[m * 4 + 1] = angle3(nix, niy, niz, dx, dy, dz);
        ppf[m * 4 + 2] = angle3(njx, njy, njz, dx, dy, dz);
        ppf[m * 4 + 3] = angle3(nix, niy, niz, njx, njy, njz);
    }
    __syncthreads();

    // build MROWSx96 bf16 message tile: [x_j(64) | ppf(4) | eattr(4) | pad(24)]
    for (int t = tid; t < MROWS * 96; t += 256) {
        int m = t / 96, f = t % 96;
        float v;
        if (f < FIN)                v = x[(size_t)jrow[m] * FIN + f];
        else if (f < FIN + 4)       v = ppf[m * 4 + (f - FIN)];
        else if (f < FIN + 4 + DE)  v = eattr[(size_t)erow[m] * DE + (f - FIN - 4)];
        else                        v = 0.0f;
        msg[t] = f2bf(v);
    }
    __syncthreads();

    // ---- layer 1: [Mx96] @ [96x128] -> relu -> hb (bf16) ----
    int col = wv * 16 + lo16;
#pragma unroll
    for (int mt = 0; mt < MTILES; ++mt) {
        F32x8 acc;
#pragma unroll
        for (int r = 0; r < 8; ++r) acc.f[r] = 0.0f;
#pragma unroll
        for (int ks = 0; ks < 3; ++ks) {
            v16bf a = load_a_frag(msg, mt * 16 + lo16, 96, ks * 32, hi);
            v16bf b = load_b_frag(w1s, col, 96, ks * 32, hi);
            acc.v = __builtin_amdgcn_wmma_f32_16x16x32_bf16(
                false, a, false, b, (short)0, acc.v, false, false);
        }
#pragma unroll
        for (int r = 0; r < 8; ++r) {
            int m = mt * 16 + r + hi * 8;
            float v = acc.f[r] + bias1[col];
            hb[m * HID + col] = f2bf(v > 0.0f ? v : 0.0f);
        }
    }
    __syncthreads();

    // ---- layer 2: [Mx128] @ [128x128] -> relu -> masked max per target ----
    float lmax0 = -__builtin_inff();
    float lmax1 = -__builtin_inff();
#pragma unroll
    for (int mt = 0; mt < MTILES; ++mt) {
        F32x8 acc;
#pragma unroll
        for (int r = 0; r < 8; ++r) acc.f[r] = 0.0f;
#pragma unroll
        for (int ks = 0; ks < 4; ++ks) {
            v16bf a = load_a_frag(hb, mt * 16 + lo16, HID, ks * 32, hi);
            v16bf b = load_b_frag(w2s, col, HID, ks * 32, hi);
            acc.v = __builtin_amdgcn_wmma_f32_16x16x32_bf16(
                false, a, false, b, (short)0, acc.v, false, false);
        }
#pragma unroll
        for (int r = 0; r < 8; ++r) {
            int m = mt * 16 + r + hi * 8;
            float v = acc.f[r] + bias2[col];
            v = v > 0.0f ? v : 0.0f;
            if (vm[m]) {
                if (mt < MTILES / 2) lmax0 = lmax0 > v ? lmax0 : v;
                else                 lmax1 = lmax1 > v ? lmax1 : v;
            }
        }
    }
    float o0 = __shfl_xor(lmax0, 16, 32);
    float o1 = __shfl_xor(lmax1, 16, 32);
    lmax0 = lmax0 > o0 ? lmax0 : o0;
    lmax1 = lmax1 > o1 ? lmax1 : o1;
    if (lane < 16) {
        agg[(size_t)(i0 + 0) * HID + col] = lmax0;
        agg[(size_t)(i0 + 1) * HID + col] = lmax1;
    }
}

// ---------------------------------------------------------------------------
// 5) global_nn: out = agg @ Wg + bg   (64 targets per block, WMMA bf16)
// ---------------------------------------------------------------------------
__global__ __launch_bounds__(256) void final_gemm(
    const float* __restrict__ agg, const unsigned short* __restrict__ WgT,
    const float* __restrict__ bg, float* __restrict__ out) {
    __shared__ unsigned short As[64 * HID];
    __shared__ unsigned short Bs[HID * FOUT];
    __shared__ float bgs[FOUT];

    int tid = threadIdx.x;
    int row0 = blockIdx.x * 64;
    int wv = tid >> 5, lane = tid & 31, lo16 = lane & 15, hi = lane >> 4;

    for (int t = tid; t < 64 * HID; t += 256)
        As[t] = f2bf(agg[(size_t)row0 * HID + t]);
    for (int t = tid; t < HID * FOUT / 2; t += 256)
        ((unsigned int*)Bs)[t] = ((const unsigned int*)WgT)[t];
    if (tid < FOUT) bgs[tid] = bg[tid];
    __syncthreads();

    int col = wv * 16 + lo16;
#pragma unroll
    for (int mt = 0; mt < 4; ++mt) {
        F32x8 acc;
#pragma unroll
        for (int r = 0; r < 8; ++r) acc.f[r] = 0.0f;
#pragma unroll
        for (int ks = 0; ks < 4; ++ks) {
            v16bf a = load_a_frag(As, mt * 16 + lo16, HID, ks * 32, hi);
            v16bf b = load_b_frag(Bs, col, HID, ks * 32, hi);
            acc.v = __builtin_amdgcn_wmma_f32_16x16x32_bf16(
                false, a, false, b, (short)0, acc.v, false, false);
        }
#pragma unroll
        for (int r = 0; r < 8; ++r) {
            int m = mt * 16 + r + hi * 8;
            out[(size_t)(row0 + m) * FOUT + col] = acc.f[r] + bgs[col];
        }
    }
}

// ---------------------------------------------------------------------------
// launch
// ---------------------------------------------------------------------------
extern "C" void kernel_launch(void* const* d_in, const int* in_sizes, int n_in,
                              void* d_out, int out_size, void* d_ws, size_t ws_size,
                              hipStream_t stream) {
    const float* x      = (const float*)d_in[0];
    const float* pos    = (const float*)d_in[1];
    const float* normal = (const float*)d_in[2];
    const int*   batch  = (const int*)d_in[3];
    const float* eattr  = (const float*)d_in[4];
    const int*   eidx2  = (const int*)d_in[5];
    const float* W1     = (const float*)d_in[6];
    const float* b1     = (const float*)d_in[7];
    const float* W2     = (const float*)d_in[8];
    const float* b2     = (const float*)d_in[9];
    const float* Wg     = (const float*)d_in[10];
    const float* bg     = (const float*)d_in[11];
    float* out = (float*)d_out;

    char* ws = (char*)d_ws;
    int*            nbr   = (int*)(ws + OFF_NBR);
    unsigned char*  valid = (unsigned char*)(ws + OFF_VALID);
    int*            eidx  = (int*)(ws + OFF_EIDX);
    unsigned int*   hkey  = (unsigned int*)(ws + OFF_HKEY);
    unsigned int*   hval  = (unsigned int*)(ws + OFF_HVAL);
    unsigned short* W1T   = (unsigned short*)(ws + OFF_W1T);
    unsigned short* W2T   = (unsigned short*)(ws + OFF_W2T);
    unsigned short* WgT   = (unsigned short*)(ws + OFF_WGT);
    float*          agg   = (float*)(ws + OFF_AGG);

    prep_weights<<<1, 256, 0, stream>>>(W1, W2, Wg, W1T, W2T, WgT);
    init_hash<<<HSZ / 256, 256, 0, stream>>>(hkey, hval);
    build_hash<<<EG / 256, 256, 0, stream>>>(eidx2, hkey, hval);
    neighbors_kernel<<<NPTS, 256, 0, stream>>>(pos, batch, nbr, valid);
    edge_lookup<<<NPTS * KNBR / 256, 256, 0, stream>>>(nbr, eidx, hkey, hval);

    size_t mlp_lds = (size_t)(HID * 96 + HID * HID + MROWS * 96 + MROWS * HID) * 2
                   + (size_t)(MROWS * 4 + 2 * HID) * 4
                   + (size_t)(3 * MROWS) * 4;
    mlp_kernel<<<NPTS / PPB, 256, mlp_lds, stream>>>(x, pos, normal, eattr, nbr, valid,
                                                     eidx, W1T, W2T, b1, b2, agg);
    final_gemm<<<NPTS / 64, 256, 0, stream>>>(agg, WgT, bg, out);

    // tuple return: (out, pos, batch) concatenated flat
    size_t outElems = (size_t)NPTS * FOUT;
    size_t posElems = (size_t)NPTS * 3;
    hipMemcpyAsync(out + outElems, pos, posElems * sizeof(float),
                   hipMemcpyDeviceToDevice, stream);
    long long remElems = (long long)out_size - (long long)(outElems + posElems);
    if (remElems > 0) {
        hipMemcpyAsync(out + outElems + posElems, batch,
                       (size_t)remElems * sizeof(float),
                       hipMemcpyDeviceToDevice, stream);
    }
}
